// LiftSplatShootBEVGenerator_8864812499672
// MI455X (gfx1250) — compile-verified, matrix operand order checked by reading, not run on hardware
//
#include <hip/hip_runtime.h>

// ---------------- problem constants (from reference) ----------------
#define B_     4
#define N_     6
#define CIN_   256
#define C_     64
#define D_     41
#define FH_    16
#define FW_    44
#define HWF    (FH_*FW_)        // 704
#define NIMG   (B_*N_)          // 24
#define NPIX_F (NIMG*HWF)       // 16896  = 66*256
#define NX0_   200
#define NX1_   200
#define HWB    (NX0_*NX1_)      // 40000
#define NPIX_B (B_*HWB)         // 160000 = 625*256

typedef __attribute__((ext_vector_type(16))) __bf16        v16bf;
typedef __attribute__((ext_vector_type(8)))  float         v8f;
typedef __attribute__((ext_vector_type(8)))  unsigned int  v8u;
typedef __attribute__((ext_vector_type(4)))  unsigned int  u32x4;
typedef __attribute__((ext_vector_type(8)))  int           i32x8;
typedef __attribute__((ext_vector_type(4)))  int           i32x4;

#ifndef __has_builtin
#define __has_builtin(x) 0
#endif
#if __has_builtin(__builtin_amdgcn_tensor_load_to_lds) && __has_builtin(__builtin_amdgcn_s_wait_tensorcnt)
#define USE_TDM 1
#else
#define USE_TDM 0
#endif

// fp32 -> bf16 round-to-nearest-even
__device__ __forceinline__ unsigned short f2bf(float f) {
    unsigned int u = __builtin_bit_cast(unsigned int, f);
    u += 0x7FFFu + ((u >> 16) & 1u);
    return (unsigned short)(u >> 16);
}

__global__ __launch_bounds__(256)
void cvt_bf16(const float* __restrict__ s, unsigned short* __restrict__ d, int n)
{
    int i = blockIdx.x * 256 + threadIdx.x;
    if (i < n) d[i] = f2bf(s[i]);
}

// =====================================================================
// Implicit-GEMM convolution on the WMMA pipe (all operands bf16, f32 acc).
//   GEMM: M = Cout, N = pixels, K = Cin*KH*KW (multiple of 32 for all layers)
//   Block tile 64(M) x 256(N); 8 wave32 waves laid out 2(M) x 4(N);
//   wave tile 32x64 -> 8x v_wmma_f32_16x16x32_bf16 per K step.
//   A tile (bf16 weights) DMA'd by the Tensor Data Mover into LDS with
//   descriptor padding (row stride 40 shorts = conflict-free b32 reads);
//   B tile (im2col gather) staged by all 256 threads while the TDM runs.
// =====================================================================
template<int KH, int KW, int PAD, bool RELU, bool OUTF>
__global__ __launch_bounds__(256)
void conv_wmma(const unsigned short* __restrict__ in, const unsigned short* __restrict__ wgt,
               const float* __restrict__ bias, void* __restrict__ outp,
               int Cin, int Cout, int H, int W)
{
    const int HW   = H * W;
    const int K    = Cin * KH * KW;
    const int tid  = threadIdx.x;
    const int bm   = blockIdx.y * 64;
    const int bn   = blockIdx.x * 256;
    const int wv   = tid >> 5;
    const int wm   = wv >> 2;             // 0..1 (M half)
    const int wn   = wv & 3;              // 0..3 (N quarter)
    const int lane = tid & 31;
    const int half = lane >> 4;
    const int l16  = lane & 15;

    __shared__ unsigned short As[64 * 40];   // [m][k], stride 40 shorts (80B)
    __shared__ unsigned short Bs[256 * 40];  // [n][k], stride 40 shorts

    // one thread <-> one output column: hoist all im2col pixel math
    const int p   = bn + tid;
    const int im  = p / HW;
    const int pix = p - im * HW;
    const int oy  = pix / W;
    const int ox  = pix - oy * W;
    const long long inbase = (long long)im * Cin * HW;

    int ci = 0, rr = 0;                   // incremental k -> (ci,fy,fx) tracking
    v8f acc[2][4] = {};

    for (int k0 = 0; k0 < K; k0 += 32) {
        __syncthreads();

#if USE_TDM
        // ---- A tile: Tensor Data Mover, global(bf16 weights) -> LDS ----
        if (wv == 0) {
            unsigned long long ga = (unsigned long long)(const void*)(wgt + (long long)bm * K + k0);
            unsigned lds = (unsigned)(unsigned long long)(const void*)&As[0];
            int rows = Cout - bm;                 // OOB rows read back as zero
            u32x4 g0;
            g0[0] = 1u;                           // count=1, user mode, no gather
            g0[1] = lds;                          // lds_addr
            g0[2] = (unsigned)(ga & 0xFFFFFFFFull);
            g0[3] = (unsigned)((ga >> 32) & 0x01FFFFFFull) | (2u << 30);  // addr[56:32] | type=2
            i32x8 g1;
            g1[0] = (1 << 16)                      // data_size = 2B
                  | (1 << 20)                      // pad_enable
                  | (3 << 22)                      // pad_interval: 16 DWORDs (one 32-elem row)
                  | (3 << 25);                     // pad_amount: 4 DWORDs (-> stride 40 shorts)
            g1[1] = ((K - k0) & 0xFFFF) << 16;     // tensor_dim0[15:0]
            g1[2] = (((K - k0) >> 16) & 0xFFFF) | ((rows & 0xFFFF) << 16);  // dim0[31:16] | dim1[15:0]
            g1[3] = ((rows >> 16) & 0xFFFF) | (32 << 16);                   // dim1[31:16] | tile_dim0=32
            g1[4] = 64;                            // tile_dim1=64, tile_dim2=0
            g1[5] = K;                             // tensor_dim0_stride[31:0] (elements)
            g1[6] = 0;                             // stride[47:32] | dim1_stride lo
            g1[7] = 0;
            i32x4 gz4 = {};
            i32x8 gz8 = {};
            __builtin_amdgcn_tensor_load_to_lds(g0, g1, gz4, gz4, gz8, 0);
        }
#else
        // ---- A tile fallback: manual bf16 copy (b128 loads / b32 LDS stores) ----
        {
            int ml = tid >> 2, kb = (tid & 3) * 8;
            int gm = bm + ml;
            const unsigned int* src = (const unsigned int*)(wgt + (long long)gm * K + k0 + kb);
            #pragma unroll
            for (int j = 0; j < 4; ++j) {
                unsigned int v = (gm < Cout) ? src[j] : 0u;
                *(unsigned int*)&As[ml * 40 + kb + j * 2] = v;
            }
        }
#endif
        // ---- B tile: im2col gather (bf16), runs concurrently with the TDM ----
        #pragma unroll
        for (int kk = 0; kk < 32; kk += 2) {
            unsigned int packed = 0;
            #pragma unroll
            for (int j = 0; j < 2; ++j) {
                int fy = 0, fx = 0;
                if (KH == 3) { fy = (rr >= 6) ? 2 : ((rr >= 3) ? 1 : 0); fx = rr - fy * 3; }
                int iy = oy + fy - PAD, ix = ox + fx - PAD;
                unsigned short v = 0;
                bool ok = (PAD == 0) || (iy >= 0 && iy < H && ix >= 0 && ix < W);
                if (ok) v = in[inbase + (long long)ci * HW + iy * W + ix];
                packed |= ((unsigned int)v) << (16 * j);
                if (++rr == KH * KW) { rr = 0; ++ci; }
            }
            *(unsigned int*)&Bs[tid * 40 + kk] = packed;
        }
#if USE_TDM
        if (wv == 0) __builtin_amdgcn_s_wait_tensorcnt(0);
#endif
        __syncthreads();

        // ---- fragments + 8 WMMAs per wave ----
        v8u au[2], bu[4];
        #pragma unroll
        for (int i = 0; i < 8; ++i) {
            int kk = (i & 3) * 2 + 8 * half + 16 * (i >> 2);
            #pragma unroll
            for (int am = 0; am < 2; ++am)
                au[am][i] = *(const unsigned int*)(&As[(wm * 32 + am * 16 + l16) * 40 + kk]);
            #pragma unroll
            for (int nt = 0; nt < 4; ++nt)
                bu[nt][i] = *(const unsigned int*)(&Bs[(wn * 64 + nt * 16 + l16) * 40 + kk]);
        }
        v16bf a0 = __builtin_bit_cast(v16bf, au[0]);
        v16bf a1 = __builtin_bit_cast(v16bf, au[1]);
        #pragma unroll
        for (int nt = 0; nt < 4; ++nt) {
            v16bf b = __builtin_bit_cast(v16bf, bu[nt]);
            acc[0][nt] = __builtin_amdgcn_wmma_f32_16x16x32_bf16(false, a0, false, b, (short)0, acc[0][nt], false, false);
            acc[1][nt] = __builtin_amdgcn_wmma_f32_16x16x32_bf16(false, a1, false, b, (short)0, acc[1][nt], false, false);
        }
    }

    // ---- epilogue: bias (+ReLU); write fp32 or bf16 mirror, NCHW ----
    #pragma unroll
    for (int nt = 0; nt < 4; ++nt) {
        int n = bn + wn * 64 + nt * 16 + l16;
        int imo = n / HW, pixo = n - imo * HW;
        #pragma unroll
        for (int am = 0; am < 2; ++am) {
            #pragma unroll
            for (int r = 0; r < 8; ++r) {
                int m = bm + wm * 32 + am * 16 + r + 8 * half;
                if (m < Cout) {
                    float v = acc[am][nt][r] + bias[m];
                    if (RELU) v = v > 0.0f ? v : 0.0f;
                    long long oidx = (long long)(imo * Cout + m) * HW + pixo;
                    if (OUTF) ((float*)outp)[oidx] = v;
                    else      ((unsigned short*)outp)[oidx] = f2bf(v);
                }
            }
        }
    }
}

// ---------------- geometry setup: 24 cameras ----------------
__device__ __forceinline__ void inv3x3(const float* a, float* o) {
    float c00 = a[4]*a[8] - a[5]*a[7];
    float c01 = a[5]*a[6] - a[3]*a[8];
    float c02 = a[3]*a[7] - a[4]*a[6];
    float id  = 1.0f / (a[0]*c00 + a[1]*c01 + a[2]*c02);
    o[0]=c00*id; o[1]=(a[2]*a[7]-a[1]*a[8])*id; o[2]=(a[1]*a[5]-a[2]*a[4])*id;
    o[3]=c01*id; o[4]=(a[0]*a[8]-a[2]*a[6])*id; o[5]=(a[2]*a[3]-a[0]*a[5])*id;
    o[6]=c02*id; o[7]=(a[1]*a[6]-a[0]*a[7])*id; o[8]=(a[0]*a[4]-a[1]*a[3])*id;
}

// geo layout (floats): ipr[24][9] | comb[24][9] @216 | trans[24][3] @432 | ptrans[24][3] @504
__global__ void setup_geo(const float* __restrict__ rots, const float* __restrict__ intrins,
                          const float* __restrict__ post_rots, const float* __restrict__ trans,
                          const float* __restrict__ post_trans, float* __restrict__ geo)
{
    int q = threadIdx.x;
    if (q >= NIMG) return;
    float ipr[9], ii[9];
    inv3x3(post_rots + q * 9, ipr);
    inv3x3(intrins   + q * 9, ii);
    const float* R = rots + q * 9;
    #pragma unroll
    for (int r = 0; r < 3; ++r)
        #pragma unroll
        for (int c = 0; c < 3; ++c)
            geo[216 + q*9 + r*3 + c] = R[r*3+0]*ii[c] + R[r*3+1]*ii[3+c] + R[r*3+2]*ii[6+c];
    #pragma unroll
    for (int i = 0; i < 9; ++i) geo[q*9 + i] = ipr[i];
    #pragma unroll
    for (int i = 0; i < 3; ++i) { geo[432 + q*3 + i] = trans[q*3+i]; geo[504 + q*3 + i] = post_trans[q*3+i]; }
}

// ---------------- depth softmax over 41 bins ----------------
__global__ __launch_bounds__(256)
void softmax_depth(const float* __restrict__ logits, float* __restrict__ dst)
{
    int idx = blockIdx.x * 256 + threadIdx.x;
    if (idx >= NPIX_F) return;
    int q = idx / HWF, pix = idx - q * HWF;
    const float* src = logits + (long long)q * D_ * HWF + pix;
    float*       o   = dst    + (long long)q * D_ * HWF + pix;
    float mx = -1e30f;
    for (int d = 0; d < D_; ++d) { float v = src[d * HWF]; mx = v > mx ? v : mx; }
    float s = 0.0f;
    for (int d = 0; d < D_; ++d) { float e = __expf(src[d * HWF] - mx); o[d * HWF] = e; s += e; }
    float is = 1.0f / s;
    for (int d = 0; d < D_; ++d) o[d * HWF] *= is;
}

// ---------------- lift (outer product) + splat (scatter-add) ----------------
__global__ __launch_bounds__(256)
void lift_splat(const float* __restrict__ depth, const float* __restrict__ imgf,
                const float* __restrict__ geo, float* __restrict__ pooled)
{
    long long g = (long long)blockIdx.x * 256 + threadIdx.x;
    int c = (int)(g & 63);
    long long pt = g >> 6;
    if (pt >= (long long)NIMG * D_ * HWF) return;
    int w = (int)(pt % FW_);  long long t = pt / FW_;
    int h = (int)(t % FH_);   t /= FH_;
    int d = (int)(t % D_);    t /= D_;
    int q = (int)t;                      // b*6+n
    int b = q / N_;

    float xf = w * (703.0f / 43.0f);
    float yf = h * 17.0f;                // 255/15
    float df = 4.0f + (float)d;

    const float* ipr  = geo + q * 9;
    const float* comb = geo + 216 + q * 9;
    const float* tr   = geo + 432 + q * 3;
    const float* ptr_ = geo + 504 + q * 3;

    float px = xf - ptr_[0], py = yf - ptr_[1], pz = df - ptr_[2];
    float ux = ipr[0]*px + ipr[1]*py + ipr[2]*pz;
    float uy = ipr[3]*px + ipr[4]*py + ipr[5]*pz;
    float uz = ipr[6]*px + ipr[7]*py + ipr[8]*pz;
    ux *= uz; uy *= uz;
    float ex = comb[0]*ux + comb[1]*uy + comb[2]*uz + tr[0];
    float ey = comb[3]*ux + comb[4]*uy + comb[5]*uz + tr[1];
    float ez = comb[6]*ux + comb[7]*uy + comb[8]*uz + tr[2];

    // reference: ((pts - (bx - dx/2)) / dx).astype(int32) -> trunc toward zero
    int gx = (int)((ex + 50.0f) * 2.0f);
    int gy = (int)((ey + 50.0f) * 2.0f);
    int gz = (int)((ez + 10.0f) * 0.05f);
    if (gx < 0 || gx >= NX0_ || gy < 0 || gy >= NX1_ || gz < 0 || gz >= 1) return;

    int pix = h * FW_ + w;
    float dval = depth[(long long)(q * D_ + d) * HWF + pix];
    float ival = imgf [(long long)(q * C_ + c) * HWF + pix];
    atomicAdd(pooled + ((long long)(b * C_ + c) * NX0_ + gx) * NX1_ + gy, dval * ival);
}

__global__ void zero_f(float* __restrict__ p, long long n)
{
    long long i = (long long)blockIdx.x * 256 + threadIdx.x;
    if (i < n) p[i] = 0.0f;
}

// =====================================================================
extern "C" void kernel_launch(void* const* d_in, const int* in_sizes, int n_in,
                              void* d_out, int out_size, void* d_ws, size_t ws_size,
                              hipStream_t stream)
{
    (void)in_sizes; (void)n_in; (void)out_size; (void)ws_size;
    const float* feat       = (const float*)d_in[0];
    const float* rots       = (const float*)d_in[1];
    const float* trans      = (const float*)d_in[2];
    const float* intrins    = (const float*)d_in[3];
    const float* post_trans = (const float*)d_in[5];
    const float* post_rots  = (const float*)d_in[6];
    const float* cam_w1 = (const float*)d_in[7],  *cam_b1 = (const float*)d_in[8];
    const float* cam_w2 = (const float*)d_in[9],  *cam_b2 = (const float*)d_in[10];
    const float* cam_wl = (const float*)d_in[11], *cam_bl = (const float*)d_in[12];
    const float* dep_w1 = (const float*)d_in[13], *dep_b1 = (const float*)d_in[14];
    const float* dep_wl = (const float*)d_in[15], *dep_bl = (const float*)d_in[16];
    const float* fus_w  = (const float*)d_in[17], *fus_b  = (const float*)d_in[18];
    const float* bev_w1 = (const float*)d_in[19], *bev_b1 = (const float*)d_in[20];
    const float* bev_w2 = (const float*)d_in[21], *bev_b2 = (const float*)d_in[22];
    const float* bev_wl = (const float*)d_in[23], *bev_bl = (const float*)d_in[24];

    // ---- workspace layout (byte offsets, all 256B aligned; ~212 MB) ----
    char* wsb = (char*)d_ws;
    float*          geo     = (float*)         (wsb + 0);           //   2,304 B
    float*          img     = (float*)         (wsb + 2560);        // 4,325,376 B
    float*          depth   = (float*)         (wsb + 4327936);     // 2,770,944 B
    float*          logits  = (float*)         (wsb + 7098880);     // 2,770,944 B
    float*          pooled  = (float*)         (wsb + 9869824);     // 40,960,000 B
    unsigned short* featb   = (unsigned short*)(wsb + 50829824);    // 8,650,752 B
    unsigned short* h1b     = (unsigned short*)(wsb + 59480576);    // 8,650,752 B
    unsigned short* h2b     = (unsigned short*)(wsb + 68131328);    // 8,650,752 B
    unsigned short* d1b     = (unsigned short*)(wsb + 76782080);    // 8,650,752 B
    unsigned short* fusedb  = (unsigned short*)(wsb + 85432832);    // 20,480,000 B
    unsigned short* bh1b    = (unsigned short*)(wsb + 105912832);   // 40,960,000 B
    unsigned short* bh2b    = (unsigned short*)(wsb + 146872832);   // 40,960,000 B
    unsigned short* pooledb = (unsigned short*)(wsb + 187832832);   // 20,480,000 B
    unsigned short* wbase   = (unsigned short*)(wsb + 208312832);   // 3,011,072 B
    unsigned short* cam_w1b = wbase + 0;
    unsigned short* cam_w2b = wbase + 589824;
    unsigned short* cam_wlb = wbase + 1179648;
    unsigned short* dep_w1b = wbase + 1196032;
    unsigned short* dep_wlb = wbase + 1261568;
    unsigned short* fus_wb  = wbase + 1272064;
    unsigned short* bev_w1b = wbase + 1276160;
    unsigned short* bev_w2b = wbase + 1349888;
    unsigned short* bev_wlb = wbase + 1497344;

    setup_geo<<<1, 32, 0, stream>>>(rots, intrins, post_rots, trans, post_trans, geo);
    zero_f<<<(10240000 + 255) / 256, 256, 0, stream>>>(pooled, 10240000LL);

    // ---- one-time bf16 conversions (weights + input features) ----
    cvt_bf16<<<(4325376 + 255) / 256, 256, 0, stream>>>(feat,   featb,   4325376);
    cvt_bf16<<<(589824  + 255) / 256, 256, 0, stream>>>(cam_w1, cam_w1b, 589824);
    cvt_bf16<<<(589824  + 255) / 256, 256, 0, stream>>>(cam_w2, cam_w2b, 589824);
    cvt_bf16<<<(16384   + 255) / 256, 256, 0, stream>>>(cam_wl, cam_wlb, 16384);
    cvt_bf16<<<(65536   + 255) / 256, 256, 0, stream>>>(dep_w1, dep_w1b, 65536);
    cvt_bf16<<<(10496   + 255) / 256, 256, 0, stream>>>(dep_wl, dep_wlb, 10496);
    cvt_bf16<<<(4096    + 255) / 256, 256, 0, stream>>>(fus_w,  fus_wb,  4096);
    cvt_bf16<<<(73728   + 255) / 256, 256, 0, stream>>>(bev_w1, bev_w1b, 73728);
    cvt_bf16<<<(147456  + 255) / 256, 256, 0, stream>>>(bev_w2, bev_w2b, 147456);
    cvt_bf16<<<(8192    + 255) / 256, 256, 0, stream>>>(bev_wl, bev_wlb, 8192);

    // ---- cam encoder ----
    conv_wmma<3,3,1,true ,false><<<dim3(NPIX_F/256, 4), 256, 0, stream>>>(featb, cam_w1b, cam_b1, h1b, CIN_, 256, FH_, FW_);
    conv_wmma<3,3,1,true ,false><<<dim3(NPIX_F/256, 4), 256, 0, stream>>>(h1b,   cam_w2b, cam_b2, h2b, 256,  256, FH_, FW_);
    conv_wmma<1,1,0,false,true ><<<dim3(NPIX_F/256, 1), 256, 0, stream>>>(h2b,   cam_wlb, cam_bl, img, 256,  C_,  FH_, FW_);
    // ---- depth encoder + softmax ----
    conv_wmma<1,1,0,true ,false><<<dim3(NPIX_F/256, 4), 256, 0, stream>>>(featb, dep_w1b, dep_b1, d1b,    CIN_, 256, FH_, FW_);
    conv_wmma<1,1,0,false,true ><<<dim3(NPIX_F/256, 1), 256, 0, stream>>>(d1b,   dep_wlb, dep_bl, logits, 256,  D_,  FH_, FW_);
    softmax_depth<<<(NPIX_F + 255) / 256, 256, 0, stream>>>(logits, depth);
    // ---- lift + splat (fused outer product, segment-sum, transpose) ----
    long long npc = (long long)NIMG * D_ * HWF * 64;   // 44,335,104
    lift_splat<<<(unsigned)((npc + 255) / 256), 256, 0, stream>>>(depth, img, geo, pooled);
    cvt_bf16<<<(10240000 + 255) / 256, 256, 0, stream>>>(pooled, pooledb, 10240000);
    // ---- fusion + BEV encoder ----
    conv_wmma<1,1,0,false,false><<<dim3(NPIX_B/256, 1), 256, 0, stream>>>(pooledb, fus_wb,  fus_b,  fusedb, C_,  C_,  NX0_, NX1_);
    conv_wmma<3,3,1,true ,false><<<dim3(NPIX_B/256, 2), 256, 0, stream>>>(fusedb,  bev_w1b, bev_b1, bh1b,   C_,  128, NX0_, NX1_);
    conv_wmma<3,3,1,true ,false><<<dim3(NPIX_B/256, 2), 256, 0, stream>>>(bh1b,    bev_w2b, bev_b2, bh2b,   128, 128, NX0_, NX1_);
    conv_wmma<1,1,0,false,true ><<<dim3(NPIX_B/256, 1), 256, 0, stream>>>(bh2b,    bev_wlb, bev_bl, d_out,  128, C_,  NX0_, NX1_);
}
